// DenseSwAVCriterion_17411797418286
// MI455X (gfx1250) — matrix operand order repaired and checked by reading, not compiled, not run on hardware
//
#include <hip/hip_runtime.h>
#include <math.h>

// ---------------- problem constants ----------------
constexpr int MN_ = 8;      // M*N images
constexpr int P_  = 3136;   // pixels per image
constexpr int D_  = 128;    // proj dim
constexpr int K_  = 3000;   // clusters
constexpr int R_  = 64;     // regions per image
constexpr int NS_ = 256;    // N*R samples per view
constexpr float EPSILON_   = 0.05f;
constexpr float TEMP_      = 0.1f;
constexpr float PDIST_EPS_ = 1e-6f;

// ---------------- workspace layout (floats) ----------------
constexpr size_t o_smax  = 0;                       // MN*P
constexpr size_t o_sx    = o_smax  + (size_t)MN_*P_;
constexpr size_t o_sx2   = o_sx    + (size_t)MN_*P_;
constexpr size_t o_sy    = o_sx2   + (size_t)MN_*P_;       // MN*R
constexpr size_t o_sy2   = o_sy    + (size_t)MN_*R_;
constexpr size_t o_rid   = o_sy2   + (size_t)MN_*R_;       // MN*R ints
constexpr size_t o_rproj = o_rid   + (size_t)MN_*R_;       // MN*R*D
constexpr size_t o_rsc   = o_rproj + (size_t)MN_*R_*D_;    // MN*R*K
constexpr size_t o_Q     = o_rsc   + (size_t)MN_*R_*K_;    // K*NS
constexpr size_t o_A     = o_Q     + (size_t)K_*NS_;       // NS*K
constexpr size_t o_rowf  = o_A     + (size_t)NS_*K_;       // K
constexpr size_t o_col   = o_rowf  + (size_t)K_;           // NS
constexpr size_t o_rsum  = o_col   + (size_t)NS_;          // MN*R
constexpr size_t o_rcnt  = o_rsum  + (size_t)MN_*R_;       // MN*R
constexpr size_t o_scal  = o_rcnt  + (size_t)MN_*R_;       // 8: [0]=lc [1]=swav [2]=gmax key

// ---------------- helpers ----------------
__device__ inline unsigned enc_f(float f) {
  unsigned u = __float_as_uint(f);
  return (u & 0x80000000u) ? ~u : (u | 0x80000000u);
}
__device__ inline float dec_f(unsigned k) {
  unsigned u = (k & 0x80000000u) ? (k & 0x7FFFFFFFu) : ~k;
  return __uint_as_float(u);
}
__device__ inline float wave_sum(float v) {
  for (int o = 16; o; o >>= 1) v += __shfl_xor(v, o, 32);
  return v;
}
__device__ inline float wave_max(float v) {
  for (int o = 16; o; o >>= 1) v = fmaxf(v, __shfl_xor(v, o, 32));
  return v;
}
__device__ inline float block_sum(float v, float* sbuf) {
  __syncthreads();
  int tid = threadIdx.x;
  v = wave_sum(v);
  if ((tid & 31) == 0) sbuf[tid >> 5] = v;
  __syncthreads();
  int nw = blockDim.x >> 5;
  if (tid < 32) {
    float x = (tid < nw) ? sbuf[tid] : 0.0f;
    x = wave_sum(x);
    if (tid == 0) sbuf[0] = x;
  }
  __syncthreads();
  return sbuf[0];
}
__device__ inline float block_max(float v, float* sbuf) {
  __syncthreads();
  int tid = threadIdx.x;
  v = wave_max(v);
  if ((tid & 31) == 0) sbuf[tid >> 5] = v;
  __syncthreads();
  int nw = blockDim.x >> 5;
  if (tid < 32) {
    float x = (tid < nw) ? sbuf[tid] : -INFINITY;
    x = wave_max(x);
    if (tid == 0) sbuf[0] = x;
  }
  __syncthreads();
  return sbuf[0];
}

// ---------------- K: zero fill ----------------
__global__ void k_zero(float* p, int n) {
  int t = blockIdx.x * blockDim.x + threadIdx.x;
  if (t < n) p[t] = 0.0f;
}

// ---------------- K1: streaming row-max over scores (bandwidth bound) ----------------
__global__ void k_rowmax(const float* __restrict__ scores, float* __restrict__ smax) {
  int wave = (int)((blockIdx.x * blockDim.x + threadIdx.x) >> 5);
  int lane = threadIdx.x & 31;
  if (wave >= MN_ * P_) return;
  const float4* row = (const float4*)(scores + (size_t)wave * K_);
  float m = -INFINITY;
  for (int i = lane; i < K_ / 4; i += 32) {
    __builtin_prefetch(row + i + 256, 0, 0);   // global_prefetch_b8, speculative
    float4 v = row[i];
    m = fmaxf(m, fmaxf(fmaxf(v.x, v.y), fmaxf(v.z, v.w)));
  }
  m = wave_max(m);
  if (lane == 0) smax[wave] = m;
}

// ---------------- K2: per-image segment max + first-argmax index ----------------
__global__ void k_segmax(const float* __restrict__ smax, const int* __restrict__ region,
                         int* __restrict__ rep_idx) {
  __shared__ unsigned keys[R_];
  __shared__ unsigned idxs[R_];
  int img = blockIdx.x, tid = threadIdx.x;
  if (tid < R_) { keys[tid] = 0u; idxs[tid] = 0xFFFFFFFFu; }
  __syncthreads();
  for (int p = tid; p < P_; p += blockDim.x) {
    int r = region[img * P_ + p];
    atomicMax(&keys[r], enc_f(smax[img * P_ + p]));
  }
  __syncthreads();
  for (int p = tid; p < P_; p += blockDim.x) {
    int r = region[img * P_ + p];
    if (enc_f(smax[img * P_ + p]) == keys[r]) atomicMin(&idxs[r], (unsigned)p);
  }
  __syncthreads();
  if (tid < R_) {
    unsigned v = idxs[tid];
    rep_idx[img * R_ + tid] = (v == 0xFFFFFFFFu) ? 0 : (int)v;
  }
}

// ---------------- K3: gather reps + rep row stats + rep score rows ----------------
__global__ void k_repgather(const float* __restrict__ projs, const float* __restrict__ scores,
                            const int* __restrict__ rep_idx,
                            float* __restrict__ rep_proj, float* __restrict__ rep_score,
                            float* __restrict__ sy, float* __restrict__ sy2) {
  __shared__ float sbuf[64];
  int g = blockIdx.x;              // 0..MN*R-1
  int img = g / R_, tid = threadIdx.x;
  int p = rep_idx[g];
  const float* prow = projs + ((size_t)img * P_ + p) * D_;
  float x = (tid < D_) ? prow[tid] : 0.0f;
  if (tid < D_) rep_proj[(size_t)g * D_ + tid] = x;
  float s1 = block_sum(x, sbuf);
  float s2 = block_sum(x * x, sbuf);
  if (tid == 0) { sy[g] = s1; sy2[g] = s2; }
  const float* srow = scores + ((size_t)img * P_ + p) * K_;
  for (int k = tid; k < K_; k += blockDim.x)
    rep_score[(size_t)g * K_ + k] = srow[k];
}

// ---------------- K4: per-pixel proj row stats ----------------
__global__ void k_projstats(const float* __restrict__ projs,
                            float* __restrict__ sx, float* __restrict__ sx2) {
  int wave = (int)((blockIdx.x * blockDim.x + threadIdx.x) >> 5);
  int lane = threadIdx.x & 31;
  if (wave >= MN_ * P_) return;
  const float4* row = (const float4*)(projs + (size_t)wave * D_);
  float4 v = row[lane];
  float s1 = v.x + v.y + v.z + v.w;
  float s2 = v.x * v.x + v.y * v.y + v.z * v.z + v.w * v.w;
  s1 = wave_sum(s1);
  s2 = wave_sum(s2);
  if (lane == 0) { sx[wave] = s1; sx2[wave] = s2; }
}

// ---------------- K5: WMMA Gram (proj x rep_proj^T) fused with coherence distance ----------------
typedef _Float16 v16h __attribute__((ext_vector_type(16)));
typedef float    v8f  __attribute__((ext_vector_type(8)));

__global__ void k_wmma_coh(const float* __restrict__ projs, const float* __restrict__ rep_proj,
                           const int* __restrict__ region,
                           const float* __restrict__ sx, const float* __restrict__ sx2,
                           const float* __restrict__ sy, const float* __restrict__ sy2,
                           float* __restrict__ regsum, float* __restrict__ regcnt) {
  __shared__ float gt[8][16][R_];   // per-wave 16x64 Gram tile
  int w = threadIdx.x >> 5, lane = threadIdx.x & 31;
  int gw = blockIdx.x * 8 + w;      // wave id: 0 .. MN*(P/16)-1
  int img = gw / (P_ / 16);
  int tm  = gw % (P_ / 16);
  int hi = lane >> 4, l15 = lane & 15;

  int arow_i = tm * 16 + l15;
  const float* arow = projs + ((size_t)img * P_ + arow_i) * D_;

  v8f acc[4];
#pragma unroll
  for (int tr = 0; tr < 4; tr++)
#pragma unroll
    for (int i = 0; i < 8; i++) acc[tr][i] = 0.0f;

#pragma unroll
  for (int s = 0; s < 4; s++) {          // D=128 in 4 slabs of K=32
    int kb = s * 32;
    v16h a;
#pragma unroll
    for (int vv = 0; vv < 8; vv++) {     // 16-bit A layout, 16x32, contiguous K (k = 16*hi + elem)
      int k0 = kb + (hi ? 16 : 0) + 2 * vv;
      a[2 * vv]     = (_Float16)arow[k0];
      a[2 * vv + 1] = (_Float16)arow[k0 + 1];
    }
#pragma unroll
    for (int tr = 0; tr < 4; tr++) {     // R=64 in 4 column tiles of 16
      int rcol = tr * 16 + l15;
      const float* brow = rep_proj + ((size_t)img * R_ + rcol) * D_;
      int kb0 = kb + (hi ? 16 : 0);
      v16h b;
#pragma unroll
      for (int vv = 0; vv < 8; vv++) {   // 16-bit B layout, contiguous K (k = 16*hi + elem), col per lane
        b[2 * vv]     = (_Float16)brow[kb0 + 2 * vv];
        b[2 * vv + 1] = (_Float16)brow[kb0 + 2 * vv + 1];
      }
      acc[tr] = __builtin_amdgcn_wmma_f32_16x16x32_f16(
          false, a, false, b, (short)0, acc[tr], false, false);
    }
  }

  // spill 16x64 tile to LDS (C layout: VGPR i -> row i + 8*hi, col = lane&15)
#pragma unroll
  for (int tr = 0; tr < 4; tr++)
#pragma unroll
    for (int i = 0; i < 8; i++)
      gt[w][i + hi * 8][tr * 16 + l15] = acc[tr][i];
  __syncthreads();

  if (lane < 16) {
    int m = tm * 16 + lane;
    size_t ri = (size_t)img * P_ + m;
    int r = region[ri];
    float g  = gt[w][lane][r];
    float yS = sy[img * R_ + r], y2 = sy2[img * R_ + r];
    const float e = PDIST_EPS_;
    float dd = sx2[ri] + y2 + (float)D_ * e * e - 2.0f * g + 2.0f * e * (sx[ri] - yS);
    float d = sqrtf(fmaxf(dd, 0.0f));
    atomicAdd(&regsum[img * R_ + r], d);
    atomicAdd(&regcnt[img * R_ + r], 1.0f);
  }
}

// ---------------- K6: coherence loss reduce ----------------
__global__ void k_lc(const float* __restrict__ regsum, const float* __restrict__ regcnt,
                     float* __restrict__ scal) {
  __shared__ float sbuf[64];
  int tid = threadIdx.x;
  float v = 0.0f;
  for (int i = tid; i < MN_ * R_; i += blockDim.x)
    v += regsum[i] / fmaxf(regcnt[i], 1.0f);
  v = block_sum(v, sbuf);
  if (tid == 0) scal[0] = v / (float)(MN_ * R_);
}

// ---------------- K7: global max over view-0 rep scores ----------------
__global__ void k_gmax(const float* __restrict__ rep_score, unsigned* __restrict__ gkey) {
  __shared__ float sbuf[64];
  size_t t = (size_t)blockIdx.x * blockDim.x + threadIdx.x;
  size_t stride = (size_t)gridDim.x * blockDim.x;
  float m = -INFINITY;
  for (size_t i = t; i < (size_t)NS_ * K_; i += stride) m = fmaxf(m, rep_score[i]);
  m = block_max(m, sbuf);
  if (threadIdx.x == 0) atomicMax(gkey, enc_f(m));
}

// ---------------- K8: Q[k][n] = exp((v0 - max)/eps) ----------------
__global__ void k_expQ(const float* __restrict__ rep_score, const unsigned* __restrict__ gkey,
                       float* __restrict__ Q) {
  int k = blockIdx.x, n = threadIdx.x;
  float gmax = dec_f(*gkey);
  float v = rep_score[(size_t)n * K_ + k];
  Q[(size_t)k * NS_ + n] = expf((v - gmax) * (1.0f / EPSILON_));
}

// ---------------- Sinkhorn pieces ----------------
__global__ void k_rowfac(const float* __restrict__ Q, float* __restrict__ rowfac) {
  int wv = (int)((blockIdx.x * blockDim.x + threadIdx.x) >> 5);
  int lane = threadIdx.x & 31;
  if (wv >= K_) return;
  const float4* row = (const float4*)(Q + (size_t)wv * NS_);
  float4 a = row[lane], b = row[lane + 32];
  float s = a.x + a.y + a.z + a.w + b.x + b.y + b.z + b.w;
  s = wave_sum(s);
  if (lane == 0) rowfac[wv] = (1.0f / (float)K_) / fmaxf(s, 1e-38f);
}
__global__ void k_rowscale(float* __restrict__ Q, const float* __restrict__ rowfac) {
  Q[(size_t)blockIdx.x * NS_ + threadIdx.x] *= rowfac[blockIdx.x];
}
__global__ void k_colsum(const float* __restrict__ Q, float* __restrict__ colsum) {
  int n = threadIdx.x;
  float s = 0.0f;
  for (int j = 0; j < 50; j++)
    s += Q[(size_t)(blockIdx.x * 50 + j) * NS_ + n];
  atomicAdd(&colsum[n], s);
}
__global__ void k_colscale(float* __restrict__ Q, const float* __restrict__ colsum) {
  int n = threadIdx.x;
  Q[(size_t)blockIdx.x * NS_ + n] *= (1.0f / (float)NS_) / colsum[n];
}
__global__ void k_assign(const float* __restrict__ Q, const float* __restrict__ colsum,
                         float* __restrict__ A) {
  int k = blockIdx.x, n = threadIdx.x;
  A[(size_t)n * K_ + k] = Q[(size_t)k * NS_ + n] / colsum[n];
}

// ---------------- K: cross-entropy with view-1 log-softmax ----------------
__global__ void k_ce(const float* __restrict__ rep_score, const float* __restrict__ A,
                     float* __restrict__ scal) {
  __shared__ float vs[K_];
  __shared__ float sbuf[64];
  int n = blockIdx.x, tid = threadIdx.x;
  const float* row = rep_score + (size_t)(NS_ + n) * K_;  // view m=1 rows
  for (int k = tid; k < K_; k += blockDim.x) vs[k] = row[k] * (1.0f / TEMP_);
  __syncthreads();
  float m = -INFINITY;
  for (int k = tid; k < K_; k += blockDim.x) m = fmaxf(m, vs[k]);
  m = block_max(m, sbuf);
  float se = 0.0f;
  for (int k = tid; k < K_; k += blockDim.x) se += expf(vs[k] - m);
  se = block_sum(se, sbuf);
  float logZ = m + logf(se);
  float dot = 0.0f;
  const float* an = A + (size_t)n * K_;
  for (int k = tid; k < K_; k += blockDim.x) dot += an[k] * (vs[k] - logZ);
  dot = block_sum(dot, sbuf);
  if (tid == 0) atomicAdd(&scal[1], -dot / (float)NS_);
}

__global__ void k_final(const float* __restrict__ scal, float* __restrict__ out) {
  if (threadIdx.x == 0) out[0] = 0.5f * (scal[0] + scal[1]);
}

// ---------------- launch ----------------
extern "C" void kernel_launch(void* const* d_in, const int* in_sizes, int n_in,
                              void* d_out, int out_size, void* d_ws, size_t ws_size,
                              hipStream_t stream) {
  const float* projs  = (const float*)d_in[0];
  const float* scores = (const float*)d_in[1];
  const int*   region = (const int*)d_in[2];
  float* ws = (float*)d_ws;

  float* smax   = ws + o_smax;
  float* sx     = ws + o_sx;
  float* sx2    = ws + o_sx2;
  float* sy     = ws + o_sy;
  float* sy2    = ws + o_sy2;
  int*   rid    = (int*)(ws + o_rid);
  float* rproj  = ws + o_rproj;
  float* rsc    = ws + o_rsc;
  float* Q      = ws + o_Q;
  float* A      = ws + o_A;
  float* rowf   = ws + o_rowf;
  float* col    = ws + o_col;
  float* rsum   = ws + o_rsum;
  float* rcnt   = ws + o_rcnt;
  float* scal   = ws + o_scal;
  unsigned* gkey = (unsigned*)(scal + 2);

  // zero accumulators (rsum | rcnt | scal are contiguous)
  k_zero<<<dim3((MN_ * R_ * 2 + 8 + 255) / 256), dim3(256), 0, stream>>>(rsum, MN_ * R_ * 2 + 8);

  // stage 1: stream scores once for per-row max (bandwidth bound, ~301 MB)
  k_rowmax<<<dim3(MN_ * P_ / 8), dim3(256), 0, stream>>>(scores, smax);
  // stage 2: region argmax per image
  k_segmax<<<dim3(MN_), dim3(256), 0, stream>>>(smax, region, rid);
  // stage 3: gather representatives + stats
  k_repgather<<<dim3(MN_ * R_), dim3(256), 0, stream>>>(projs, scores, rid, rproj, rsc, sy, sy2);
  k_projstats<<<dim3(MN_ * P_ / 8), dim3(256), 0, stream>>>(projs, sx, sx2);
  // stage 4: WMMA Gram + fused coherence distances
  k_wmma_coh<<<dim3(MN_ * (P_ / 16) / 8), dim3(256), 0, stream>>>(
      projs, rproj, region, sx, sx2, sy, sy2, rsum, rcnt);
  k_lc<<<dim3(1), dim3(256), 0, stream>>>(rsum, rcnt, scal);

  // stage 5: SwAV assignments (Sinkhorn on 3000x256)
  k_gmax<<<dim3(256), dim3(256), 0, stream>>>(rsc, gkey);
  k_expQ<<<dim3(K_), dim3(NS_), 0, stream>>>(rsc, gkey, Q);
  for (int it = 0; it < 3; it++) {
    k_rowfac<<<dim3(K_ / 8), dim3(256), 0, stream>>>(Q, rowf);
    k_rowscale<<<dim3(K_), dim3(NS_), 0, stream>>>(Q, rowf);
    k_zero<<<dim3(1), dim3(256), 0, stream>>>(col, NS_);
    k_colsum<<<dim3(K_ / 50), dim3(NS_), 0, stream>>>(Q, col);
    k_colscale<<<dim3(K_), dim3(NS_), 0, stream>>>(Q, col);
  }
  k_zero<<<dim3(1), dim3(256), 0, stream>>>(col, NS_);
  k_colsum<<<dim3(K_ / 50), dim3(NS_), 0, stream>>>(Q, col);
  k_assign<<<dim3(K_), dim3(NS_), 0, stream>>>(Q, col, A);

  // stage 6: cross-entropy loss + final combine
  k_ce<<<dim3(NS_), dim3(256), 0, stream>>>(rsc, A, scal);
  k_final<<<dim3(1), dim3(64), 0, stream>>>(scal, (float*)d_out);
}
